// Attention_76433238000188
// MI455X (gfx1250) — matrix operand order repaired and checked by reading, not compiled, not run on hardware
//
#include <hip/hip_runtime.h>

typedef _Float16 half_t;
typedef __attribute__((ext_vector_type(16))) _Float16 v16h;
typedef __attribute__((ext_vector_type(8)))  _Float16 v8h;
typedef __attribute__((ext_vector_type(8)))  float    v8f;
typedef __attribute__((ext_vector_type(4)))  float    v4f;
typedef __attribute__((ext_vector_type(4)))  int      v4i;

union Frag { v16h v; v8h h[2]; };

#define WMMA16(a, b, c) \
  __builtin_amdgcn_wmma_f32_16x16x32_f16(false, (a), false, (b), (short)0, (c), false, false)

// ---------------------------------------------------------------------------
// CDNA5 async global->LDS copy (ASYNCcnt-tracked), with sync fallback.
// Probe round 3: builtin exists; param0 = AS1 v4i*, param1 = AS3 v4i*.
// ---------------------------------------------------------------------------
#if defined(__has_builtin)
#if __has_builtin(__builtin_amdgcn_global_load_async_to_lds_b128)
#define HAVE_ASYNC_LDS 1
#endif
#endif
#ifndef HAVE_ASYNC_LDS
#define HAVE_ASYNC_LDS 0
#endif

typedef __attribute__((address_space(1))) v4i* gv4i_p;
typedef __attribute__((address_space(3))) v4i* lv4i_p;

__device__ __forceinline__ void async_copy16(const half_t* g, half_t* l) {
#if HAVE_ASYNC_LDS
  __builtin_amdgcn_global_load_async_to_lds_b128((gv4i_p)g, (lv4i_p)l, 0, 0);
#else
  *(v8h*)l = *(const v8h*)g;
#endif
}

__device__ __forceinline__ void wait_async0() {
#if HAVE_ASYNC_LDS
#if __has_builtin(__builtin_amdgcn_s_wait_asynccnt)
  __builtin_amdgcn_s_wait_asynccnt(0);
#else
  asm volatile("s_wait_asynccnt 0x0" ::: "memory");
#endif
#endif
}

// ---------------------------------------------------------------------------
// f32 -> f16 conversion, 4 elements / thread (n must be a multiple of 4)
// ---------------------------------------------------------------------------
__global__ void f32_to_f16_vec_kernel(const float* __restrict__ in,
                                      half_t* __restrict__ out, int n4) {
  int i = blockIdx.x * blockDim.x + threadIdx.x;
  if (i < n4) {
    v4f x = *(const v4f*)(in + i * 4);
    half_t* o = out + i * 4;
    o[0] = (half_t)x.x; o[1] = (half_t)x.y; o[2] = (half_t)x.z; o[3] = (half_t)x.w;
  }
}

// ---------------------------------------------------------------------------
// f32 [K,N] -> f16 transposed [N,K], 32x32 LDS tiles, slabs via blockIdx.z.
// ---------------------------------------------------------------------------
__global__ __launch_bounds__(256)
void transpose_f32_to_f16_kernel(const float* __restrict__ in,
                                 half_t* __restrict__ out, int K, int N) {
  __shared__ float t[32][33];
  const size_t slab = (size_t)K * N * blockIdx.z;
  in  += slab;
  out += slab;
  int n0 = blockIdx.x * 32, k0 = blockIdx.y * 32;
  int tx = threadIdx.x & 31, ty = threadIdx.x >> 5;
  for (int it = 0; it < 4; ++it)
    t[ty + it * 8][tx] = in[(size_t)(k0 + ty + it * 8) * N + (n0 + tx)];
  __syncthreads();
  for (int it = 0; it < 4; ++it)
    out[(size_t)(n0 + ty + it * 8) * K + (k0 + tx)] = (half_t)t[tx][ty + it * 8];
}

// ---------------------------------------------------------------------------
// WMMA GEMM:  C[M,N] = act( sum_s shift_s(A)[M,K] * W_s[K,N] + bias )
//   WT = pre-transposed weights, S slabs of [N,K] row-major.
//   256 threads = 8 waves; block tile 128(M) x 128(N); wave tile 32x64 (2x4).
//   K-panel of 64 staged in LDS (async) per barrier. K%64==0, M%128==0.
// ---------------------------------------------------------------------------
__global__ __launch_bounds__(256)
void gemm_wmma_kernel(const half_t* __restrict__ A, const half_t* __restrict__ WT,
                      const float* __restrict__ bias, void* __restrict__ Cout,
                      int M, int N, int K, int S, int T, int relu, int out_f16) {
  __shared__ half_t lA[128][72];  // [m][k], 64 halves + 8 pad
  __shared__ half_t lB[128][72];  // [n][k]

  const int tid   = threadIdx.x;
  const int wave  = tid >> 5;
  const int lane  = tid & 31;
  const int khalf = lane >> 4;
  const int l16   = lane & 15;
  const int wm = wave >> 1;       // 0..3  (M direction, 32 rows each)
  const int wn = wave & 1;        // 0..1  (N direction, 64 cols each)
  const int m0 = blockIdx.y * 128;
  const int n0 = blockIdx.x * 128;

  // staging coords: 2 threads per row, 32 halves each
  const int sr = tid >> 1;
  const int sc = (tid & 1) * 32;

  v8f acc[2][4] = {};

  for (int s = 0; s < S; ++s) {
    const half_t* Ws = WT + (size_t)s * K * N;
    const int shift = s - (S - 1);   // 0 for dense, -2..0 for conv
    for (int kt = 0; kt < K; kt += 64) {
      __syncthreads();
      // ---- stage A panel: 128 rows x 64 halves (causal shift / zero pad)
      {
        int m = m0 + sr;
        int b = m / T;
        int tl = (m - b * T) + shift;
        if (tl >= 0) {
          const half_t* src = A + (size_t)(b * T + tl) * K + kt + sc;
          async_copy16(src,      &lA[sr][sc]);
          async_copy16(src + 8,  &lA[sr][sc + 8]);
          async_copy16(src + 16, &lA[sr][sc + 16]);
          async_copy16(src + 24, &lA[sr][sc + 24]);
        } else {
          v8h z = {};
          *(v8h*)&lA[sr][sc]      = z;
          *(v8h*)&lA[sr][sc + 8]  = z;
          *(v8h*)&lA[sr][sc + 16] = z;
          *(v8h*)&lA[sr][sc + 24] = z;
        }
      }
      // ---- stage B panel: 128 n-rows x 64 halves from transposed weights
      {
        int n = n0 + sr;
        if (n < N) {
          const half_t* src = Ws + (size_t)n * K + kt + sc;
          async_copy16(src,      &lB[sr][sc]);
          async_copy16(src + 8,  &lB[sr][sc + 8]);
          async_copy16(src + 16, &lB[sr][sc + 16]);
          async_copy16(src + 24, &lB[sr][sc + 24]);
        } else {
          v8h z = {};
          *(v8h*)&lB[sr][sc]      = z;
          *(v8h*)&lB[sr][sc + 8]  = z;
          *(v8h*)&lB[sr][sc + 16] = z;
          *(v8h*)&lB[sr][sc + 24] = z;
        }
      }
      wait_async0();
      __syncthreads();
      // ---- 2 K-steps of 32, 8 WMMAs each
      for (int kk = 0; kk < 64; kk += 32) {
        Frag aF[2], bF[4];
        for (int i = 0; i < 2; ++i) {
          int mrow = wm * 32 + i * 16 + l16;
          aF[i].h[0] = *(const v8h*)&lA[mrow][kk + khalf * 8];
          aF[i].h[1] = *(const v8h*)&lA[mrow][kk + 16 + khalf * 8];
        }
        for (int j = 0; j < 4; ++j) {
          int ncol = wn * 64 + j * 16 + l16;
          bF[j].h[0] = *(const v8h*)&lB[ncol][kk + khalf * 8];
          bF[j].h[1] = *(const v8h*)&lB[ncol][kk + 16 + khalf * 8];
        }
        for (int i = 0; i < 2; ++i)
          for (int j = 0; j < 4; ++j)
            acc[i][j] = WMMA16(aF[i].v, bF[j].v, acc[i][j]);
      }
    }
  }

  // ---- epilogue: bias, optional relu, f16 or f32 store
  for (int i = 0; i < 2; ++i)
    for (int j = 0; j < 4; ++j) {
      int n = n0 + wn * 64 + j * 16 + l16;
      if (n >= N) continue;
      float bv = bias ? bias[n] : 0.0f;
      for (int r = 0; r < 8; ++r) {
        int m = m0 + wm * 32 + i * 16 + r + 8 * khalf;
        float val = acc[i][j][r] + bv;
        if (relu) val = fmaxf(val, 0.0f);
        if (out_f16) ((half_t*)Cout)[(size_t)m * N + n] = (half_t)val;
        else         ((float*)Cout)[(size_t)m * N + n]  = val;
      }
    }
}

// ---------------------------------------------------------------------------
// Flash attention, causal, head dim 64.
// grid = (B*H, T/128); block = 256 threads = 8 waves; wave owns 16 query rows
// (128 q-rows per block, so each staged 64-key K/V tile is reused 2x more).
// q/k/v/o are f16 [B*T, H*64] row-major.
// ---------------------------------------------------------------------------
__global__ __launch_bounds__(256)
void attn_wmma_kernel(const half_t* __restrict__ Q, const half_t* __restrict__ Kh,
                      const half_t* __restrict__ V, half_t* __restrict__ O,
                      int T, int H) {
  __shared__ half_t lK[64][72];    // [key][d]
  __shared__ half_t lV[64][72];    // transposed: [dv][key]
  __shared__ half_t lP[128][72];   // per-wave 16-row slabs, [m][key]

  const int bh = blockIdx.x;
  const int b  = bh / H;
  const int h  = bh - b * H;
  const int iq = blockIdx.y;           // q block index (128 rows)
  const int tid   = threadIdx.x;
  const int wave  = tid >> 5;
  const int lane  = tid & 31;
  const int khalf = lane >> 4;
  const int l16   = lane & 15;

  const size_t rowstride = (size_t)H * 64;
  const size_t base = (size_t)b * T * rowstride + (size_t)h * 64;
  const int q0  = iq * 128;
  const int qw0 = q0 + wave * 16;      // first q row of this wave
  const int jmax = 2 * iq + 1;         // last 64-key block needed by this block

  // preload Q A-fragments (16 rows x 64 d = 2 k-chunks)
  Frag qF[2];
  {
    const half_t* qrow = Q + base + (size_t)(qw0 + l16) * rowstride;
    for (int kk = 0; kk < 2; ++kk) {
      qF[kk].h[0] = *(const v8h*)(qrow + kk * 32 + khalf * 8);
      qF[kk].h[1] = *(const v8h*)(qrow + kk * 32 + 16 + khalf * 8);
    }
  }

  v8f o[4] = {};
  float mrun[8], lrun[8];
  for (int r = 0; r < 8; ++r) { mrun[r] = -3.0e38f; lrun[r] = 0.0f; }

  for (int j = 0; j <= jmax; ++j) {
    __syncthreads();
    // ---- stage K (async, row-major) and V (transposed) for keys [64j, 64j+63]
    {
      int cr = tid >> 3;            // 0..31
      int cc = (tid & 7) * 8;       // 0,8,..,56
      for (int it = 0; it < 2; ++it) {
        int key = it * 32 + cr;
        const half_t* srck = Kh + base + (size_t)(j * 64 + key) * rowstride + cc;
        async_copy16(srck, &lK[key][cc]);
        const half_t* srcv = V + base + (size_t)(j * 64 + key) * rowstride + cc;
        v8h vv = *(const v8h*)srcv;
        for (int e = 0; e < 8; ++e) lV[cc + e][key] = vv[e];
      }
      wait_async0();
    }
    // prefetch next tile while this one is consumed
    if (j < jmax) {
      int r = tid & 63;
      const half_t* nx = (tid < 128 ? Kh : V) +
                         base + (size_t)((j + 1) * 64 + r) * rowstride;
      __builtin_prefetch(nx, 0, 1);
    }
    __syncthreads();

    const bool active = (j * 64 <= qw0 + 15);   // any unmasked element?
    if (active) {
      // ---- S = (Q K^T) * 1/sqrt(64)
      v8f s[4];
      for (int nt = 0; nt < 4; ++nt) {
        v8f a = {};
        for (int kk = 0; kk < 2; ++kk) {
          Frag bF;
          int keyl = nt * 16 + l16;
          bF.h[0] = *(const v8h*)&lK[keyl][kk * 32 + khalf * 8];
          bF.h[1] = *(const v8h*)&lK[keyl][kk * 32 + 16 + khalf * 8];
          a = WMMA16(qF[kk].v, bF.v, a);
        }
        s[nt] = a;
      }
      const float scale = 0.125f;
      const bool needmask = (j * 64 + 63 > qw0);
      for (int nt = 0; nt < 4; ++nt)
        for (int r = 0; r < 8; ++r) {
          float xv = s[nt][r] * scale;
          if (needmask) {
            int qi = qw0 + r + 8 * khalf;
            int ki = j * 64 + nt * 16 + l16;
            if (ki > qi) xv = -1.0e9f;
          }
          s[nt][r] = xv;
        }

      // ---- online softmax per row (8 rows per lane; row = r + 8*khalf)
      for (int r = 0; r < 8; ++r) {
        float mx = fmaxf(fmaxf(s[0][r], s[1][r]), fmaxf(s[2][r], s[3][r]));
        for (int off = 1; off < 16; off <<= 1)
          mx = fmaxf(mx, __shfl_xor(mx, off, 32));
        float mnew  = fmaxf(mrun[r], mx);
        float alpha = __expf(mrun[r] - mnew);
        mrun[r] = mnew;
        float sum = 0.0f;
        for (int nt = 0; nt < 4; ++nt) {
          float pv = __expf(s[nt][r] - mnew);
          s[nt][r] = pv;
          sum += pv;
        }
        for (int off = 1; off < 16; off <<= 1)
          sum += __shfl_xor(sum, off, 32);
        lrun[r] = lrun[r] * alpha + sum;
        for (int dt = 0; dt < 4; ++dt) o[dt][r] *= alpha;
      }

      // ---- P through LDS (C-layout -> A-fragment layout)
      for (int nt = 0; nt < 4; ++nt)
        for (int r = 0; r < 8; ++r)
          lP[wave * 16 + r + 8 * khalf][nt * 16 + l16] = (half_t)s[nt][r];
    }
    __syncthreads();
    if (active) {
      // ---- O += P V
      for (int dt = 0; dt < 4; ++dt) {
        for (int kk = 0; kk < 2; ++kk) {
          Frag aP, bV;
          int mrow = wave * 16 + l16;
          aP.h[0] = *(const v8h*)&lP[mrow][kk * 32 + khalf * 8];
          aP.h[1] = *(const v8h*)&lP[mrow][kk * 32 + 16 + khalf * 8];
          int dv = dt * 16 + l16;
          bV.h[0] = *(const v8h*)&lV[dv][kk * 32 + khalf * 8];
          bV.h[1] = *(const v8h*)&lV[dv][kk * 32 + 16 + khalf * 8];
          o[dt] = WMMA16(aP.v, bV.v, o[dt]);
        }
      }
    }
  }

  // ---- normalize and store O (f16)
  for (int dt = 0; dt < 4; ++dt)
    for (int r = 0; r < 8; ++r) {
      int qi = qw0 + r + 8 * khalf;
      float val = o[dt][r] / lrun[r];
      O[base + (size_t)qi * rowstride + dt * 16 + l16] = (half_t)val;
    }
}

// ---------------------------------------------------------------------------
// Host launcher
// ---------------------------------------------------------------------------
extern "C" void kernel_launch(void* const* d_in, const int* in_sizes, int n_in,
                              void* d_out, int out_size, void* d_ws, size_t ws_size,
                              hipStream_t stream) {
  const int B = 4, T = 2048, D = 512, H = 8, DV = 64, FS = 3;
  const int M = B * T;

  const float* x  = (const float*)d_in[0];
  const float* Wq = (const float*)d_in[1];
  const float* bq = (const float*)d_in[2];
  const float* Wk = (const float*)d_in[3];
  const float* bk = (const float*)d_in[4];
  const float* Wv = (const float*)d_in[5];
  const float* bv = (const float*)d_in[6];
  const float* W1 = (const float*)d_in[7];
  const float* b1 = (const float*)d_in[8];
  const float* W2 = (const float*)d_in[9];
  const float* b2 = (const float*)d_in[10];

  char* p = (char*)d_ws;
  auto take = [&](size_t elems) {
    half_t* r = (half_t*)p;
    p += ((elems * sizeof(half_t)) + 255) & ~(size_t)255;
    return r;
  };
  half_t* xh  = take((size_t)M * D);
  half_t* Wqh = take((size_t)FS * D * D);   // transposed slabs [s][N=512][K=512]
  half_t* Wkh = take((size_t)FS * D * D);
  half_t* Wvh = take((size_t)D * D);        // [N=512][K=512]
  half_t* W1h = take((size_t)D * DV);       // [N=64][K=512]
  half_t* W2h = take((size_t)DV * DV);      // [N=64][K=64]
  half_t* qh  = take((size_t)M * D);
  half_t* kh  = take((size_t)M * D);
  half_t* vh  = take((size_t)M * D);
  half_t* oh  = take((size_t)M * D);
  half_t* hh  = take((size_t)M * DV);

  // x: straight f32 -> f16, vectorized x4
  {
    int n4 = (M * D) / 4;
    f32_to_f16_vec_kernel<<<dim3((n4 + 255) / 256), dim3(256), 0, stream>>>(x, xh, n4);
  }
  // weights: f32 [K,N] -> f16 transposed [N,K]
  transpose_f32_to_f16_kernel<<<dim3(D / 32, D / 32, FS), dim3(256), 0, stream>>>(Wq, Wqh, D, D);
  transpose_f32_to_f16_kernel<<<dim3(D / 32, D / 32, FS), dim3(256), 0, stream>>>(Wk, Wkh, D, D);
  transpose_f32_to_f16_kernel<<<dim3(D / 32, D / 32, 1),  dim3(256), 0, stream>>>(Wv, Wvh, D, D);
  transpose_f32_to_f16_kernel<<<dim3(DV / 32, D / 32, 1), dim3(256), 0, stream>>>(W1, W1h, D, DV);
  transpose_f32_to_f16_kernel<<<dim3(DV / 32, DV / 32, 1), dim3(256), 0, stream>>>(W2, W2h, DV, DV);

  // q = causal_conv(x, Wq) + bq ; k likewise ; v = x @ Wv + bv   (all f16 out)
  gemm_wmma_kernel<<<dim3(D / 128, M / 128), dim3(256), 0, stream>>>(
      xh, Wqh, bq, qh, M, D, D, FS, T, 0, 1);
  gemm_wmma_kernel<<<dim3(D / 128, M / 128), dim3(256), 0, stream>>>(
      xh, Wkh, bk, kh, M, D, D, FS, T, 0, 1);
  gemm_wmma_kernel<<<dim3(D / 128, M / 128), dim3(256), 0, stream>>>(
      xh, Wvh, bv, vh, M, D, D, 1, T, 0, 1);

  // causal flash attention over 8 heads of dim 64 (128 q-rows per block)
  attn_wmma_kernel<<<dim3(B * H, T / 128), dim3(256), 0, stream>>>(
      qh, kh, vh, oh, T, H);

  // hidden = relu(O @ W1 + b1)  (f16 out)   N=64 -> one N-block
  gemm_wmma_kernel<<<dim3(1, M / 128), dim3(256), 0, stream>>>(
      oh, W1h, b1, hh, M, DV, D, 1, T, 1, 1);
  // out = hidden @ W2 + b2  (f32 out -> d_out)
  gemm_wmma_kernel<<<dim3(1, M / 128), dim3(256), 0, stream>>>(
      hh, W2h, b2, d_out, M, DV, DV, 1, T, 0, 0);
}
